// TinyGAT_52510270161006
// MI455X (gfx1250) — compile-verified, matrix operand order checked by reading, not run on hardware
//
#include <hip/hip_runtime.h>
#include <hip/hip_bf16.h>

// ---------------- sizes (fixed by reference) ----------------
#define BB   8
#define KK   1024
#define CC   256
#define HID  128
#define NH   4
#define OUTD 128
#define BK   (BB*KK)          // 8192 rows

typedef __attribute__((ext_vector_type(16))) __bf16    v16bf;
typedef __attribute__((ext_vector_type(8)))  float     v8f;
typedef __attribute__((ext_vector_type(4)))  unsigned  u32x4;

__device__ __forceinline__ unsigned short f2bf_bits(float f) {
    unsigned u = __builtin_bit_cast(unsigned, f);
    unsigned r = u + 0x7FFFu + ((u >> 16) & 1u);   // RNE
    return (unsigned short)(r >> 16);
}
__device__ __forceinline__ __bf16 f2bf(float f) {
    unsigned short s = f2bf_bits(f);
    return __builtin_bit_cast(__bf16, s);
}

__device__ __forceinline__ v8f wmma_bf16(v16bf a, v16bf b, v8f c) {
    return __builtin_amdgcn_wmma_f32_16x16x32_bf16(false, a, false, b, (short)0, c, false, false);
}

// 16-bit fragment load, works for both A (row0 = i-tile base, rows = M) and
// B (row0 = n-tile base, rows = N) given storage with contiguous K:
//   element j of v16bf:  K = k0 + (j/8)*16 + half*8 + (j%8),  row = row0 + (lane&15)
__device__ __forceinline__ v16bf load_frag(const __bf16* __restrict__ base,
                                           int row0, int ld, int k0, unsigned lane) {
    unsigned ln = lane & 15u, half = (lane >> 4) & 1u;
    const __bf16* p = base + (size_t)(row0 + ln) * ld + k0 + (half << 3);
    union { v16bf v; u32x4 q[2]; } f;
    f.q[0] = *(const u32x4*)p;         // K group 0: 8 bf16 (16B aligned)
    f.q[1] = *(const u32x4*)(p + 16);  // K group 1
    return f.v;
}

// A-fragment built from fp32 adjacency: (val > 0) -> 1.0bf16 else 0
__device__ __forceinline__ v16bf load_mask_frag(const float* __restrict__ rowbase,
                                                int ld, int j0, unsigned lane) {
    unsigned ln = lane & 15u, half = (lane >> 4) & 1u;
    const float* p = rowbase + (size_t)ln * ld + j0 + (half << 3);
    union { v16bf v; unsigned short us[16]; } f;
#pragma unroll
    for (int g = 0; g < 2; ++g)
#pragma unroll
        for (int t = 0; t < 8; ++t) {
            float m = p[g * 16 + t];
            f.us[g * 8 + t] = (m > 0.0f) ? (unsigned short)0x3F80u : (unsigned short)0u;
        }
    return f.v;
}

// ---------------- elementwise fp32 -> bf16 convert ----------------
__global__ void conv_bf16_kernel(const float* __restrict__ src, __bf16* __restrict__ dst, int n) {
    int i = blockIdx.x * blockDim.x + threadIdx.x;
    if (i < n) dst[i] = f2bf(src[i]);
}

// ---------------- H = Z @ Wp^T  (bf16 WMMA, fp32 out) ----------------
// Zb [8192][256] bf16 row-major, Wpb [128][256] bf16 row-major (contiguous K=C)
__global__ void proj_kernel(const __bf16* __restrict__ Zb, const __bf16* __restrict__ Wpb,
                            float* __restrict__ Hf) {
    unsigned wave = blockIdx.x * (blockDim.x >> 5) + (threadIdx.x >> 5);
    unsigned lane = threadIdx.x & 31u;
    int itile = wave >> 3;        // 512 i-tiles
    int ntile = wave & 7;         // 8 n-tiles of hid
    int i0 = itile * 16, n0 = ntile * 16;
    v8f acc = {};
#pragma unroll
    for (int k0 = 0; k0 < CC; k0 += 32) {
        v16bf a = load_frag(Zb,  i0, CC, k0, lane);
        v16bf b = load_frag(Wpb, n0, CC, k0, lane);
        acc = wmma_bf16(a, b, acc);
    }
    unsigned ln = lane & 15u, half = (lane >> 4) & 1u;
#pragma unroll
    for (int v = 0; v < 8; ++v) {
        int row = i0 + v + (int)half * 8;
        Hf[(size_t)row * HID + n0 + ln] = acc[v];
    }
}

// ---------------- per-head scalar scores h[b,k,a] ----------------
__global__ void heads_kernel(const float* __restrict__ Hf, const float* __restrict__ attn,
                             float* __restrict__ Hh) {
    int bk = blockIdx.x * blockDim.x + threadIdx.x;
    if (bk >= BK) return;
    const float* hr = Hf + (size_t)bk * HID;
    float s0 = 0, s1 = 0, s2 = 0, s3 = 0;
    for (int h = 0; h < HID; ++h) {
        float x = hr[h];
        s0 += x * attn[0 * HID + h];
        s1 += x * attn[1 * HID + h];
        s2 += x * attn[2 * HID + h];
        s3 += x * attn[3 * HID + h];
    }
    int b = bk >> 10, k = bk & (KK - 1);
    Hh[((size_t)(b * NH + 0)) * KK + k] = s0;
    Hh[((size_t)(b * NH + 1)) * KK + k] = s1;
    Hh[((size_t)(b * NH + 2)) * KK + k] = s2;
    Hh[((size_t)(b * NH + 3)) * KK + k] = s3;
}

// ---------------- e[b,a,k] = exp(h - max_k h) ----------------
__global__ void softmax_e_kernel(const float* __restrict__ Hh, float* __restrict__ e) {
    int ba = blockIdx.x;                       // 32 (b,a) pairs
    const float* h = Hh + (size_t)ba * KK;
    __shared__ float red[256];
    int t = threadIdx.x;
    float m = -3.0e38f;
    for (int k = t; k < KK; k += 256) m = fmaxf(m, h[k]);
    red[t] = m; __syncthreads();
    for (int s = 128; s > 0; s >>= 1) {
        if (t < s) red[t] = fmaxf(red[t], red[t + s]);
        __syncthreads();
    }
    float M = red[0];
    for (int k = t; k < KK; k += 256)
        e[(size_t)ba * KK + k] = __expf(h[k] - M);
}

// ---------------- V'[ba][h][k] = bf16(e[ba][k] * H[b][k][h])  (K-contiguous) ----------------
__global__ void vht_kernel(const float* __restrict__ e, const float* __restrict__ Hf,
                           __bf16* __restrict__ VhT) {
    size_t idx = (size_t)blockIdx.x * blockDim.x + threadIdx.x;   // BB*NH*HID*KK
    int k  = (int)(idx & (KK - 1));
    int h  = (int)((idx >> 10) & (HID - 1));
    int ba = (int)(idx >> 17);
    int b  = ba >> 2;
    float ev = e[(size_t)ba * KK + k];
    float hv = Hf[((size_t)(b * KK + k)) * HID + h];
    VhT[idx] = f2bf(ev * hv);
}

// ---------------- d[b,a,i] = sum_j mask(i,j) * e[b,a,j] ----------------
__global__ void denom_kernel(const float* __restrict__ Asoft, const float* __restrict__ e,
                             float* __restrict__ d) {
    int bi = blockIdx.x;                       // 8192 rows
    int b = bi >> 10, i = bi & (KK - 1);
    const float* mrow = Asoft + (size_t)bi * KK;
    const float* eb = e + (size_t)b * NH * KK;
    int t = threadIdx.x;
    float p0 = 0, p1 = 0, p2 = 0, p3 = 0;
    for (int j = t; j < KK; j += 256) {
        if (mrow[j] > 0.0f) {
            p0 += eb[j];
            p1 += eb[KK + j];
            p2 += eb[2 * KK + j];
            p3 += eb[3 * KK + j];
        }
    }
    __shared__ float red[4][256];
    red[0][t] = p0; red[1][t] = p1; red[2][t] = p2; red[3][t] = p3;
    __syncthreads();
    for (int s = 128; s > 0; s >>= 1) {
        if (t < s) {
            red[0][t] += red[0][t + s];
            red[1][t] += red[1][t + s];
            red[2][t] += red[2][t + s];
            red[3][t] += red[3][t + s];
        }
        __syncthreads();
    }
    if (t == 0) {
        d[(size_t)(b * NH + 0) * KK + i] = red[0][0];
        d[(size_t)(b * NH + 1) * KK + i] = red[1][0];
        d[(size_t)(b * NH + 2) * KK + i] = red[2][0];
        d[(size_t)(b * NH + 3) * KK + i] = red[3][0];
    }
}

// ---------------- P = (Mask/denom/4) @ V'   (the big bf16 WMMA GEMM) ----------------
__global__ void agg_kernel(const float* __restrict__ Asoft, const __bf16* __restrict__ VhT,
                           const float* __restrict__ d, float* __restrict__ Pagg) {
    unsigned wave = blockIdx.x * (blockDim.x >> 5) + (threadIdx.x >> 5);  // 2048 waves
    unsigned lane = threadIdx.x & 31u;
    int itile = wave & 63;          // 64 i-tiles of 16
    int ba = wave >> 6;             // 0..31
    int b = ba >> 2;
    int i0 = itile * 16;
    const float* mbase = Asoft + ((size_t)b * KK + i0) * KK;
    const __bf16* vb   = VhT + (size_t)ba * HID * KK;
    v8f acc[8] = {};
    for (int j0 = 0; j0 < KK; j0 += 32) {
        v16bf ma = load_mask_frag(mbase, KK, j0, lane);
#pragma unroll
        for (int nt = 0; nt < 8; ++nt) {
            v16bf bm = load_frag(vb, nt * 16, KK, j0, lane);
            acc[nt] = wmma_bf16(ma, bm, acc[nt]);
        }
    }
    unsigned ln = lane & 15u, half = (lane >> 4) & 1u;
    const float* drow = d + (size_t)ba * KK + i0;
    float sc[8];
#pragma unroll
    for (int v = 0; v < 8; ++v) {
        float dv = drow[v + (int)half * 8];
        sc[v] = (dv > 0.0f) ? (0.25f / dv) : 0.0f;   // /denominator, /4 heads
    }
    float* orow = Pagg + ((size_t)ba * KK + i0) * HID;
#pragma unroll
    for (int nt = 0; nt < 8; ++nt)
#pragma unroll
        for (int v = 0; v < 8; ++v)
            orow[(size_t)(v + half * 8) * HID + nt * 16 + ln] = acc[nt][v] * sc[v];
}

// ---------------- H_agg (mean over heads) -> bf16 ----------------
__global__ void sumheads_kernel(const float* __restrict__ Pagg, __bf16* __restrict__ Hgb) {
    size_t idx = (size_t)blockIdx.x * blockDim.x + threadIdx.x;  // BK*HID
    size_t stride = (size_t)KK * HID;                            // per-head stride
    int h = (int)(idx & (HID - 1));
    size_t bk = idx >> 7;
    size_t b = bk >> 10, k = bk & (KK - 1);
    size_t base = ((b * NH) * KK + k) * HID + h;
    float s = Pagg[base] + Pagg[base + stride] + Pagg[base + 2 * stride] + Pagg[base + 3 * stride];
    Hgb[idx] = f2bf(s);   // 0.25 already folded into agg epilogue
}

// ---------------- y = elu(H_agg @ Wo^T + bo)  (bf16 WMMA) ----------------
__global__ void out_kernel(const __bf16* __restrict__ Hgb, const __bf16* __restrict__ Wob,
                           const float* __restrict__ bo, float* __restrict__ y) {
    unsigned wave = blockIdx.x * (blockDim.x >> 5) + (threadIdx.x >> 5);
    unsigned lane = threadIdx.x & 31u;
    int itile = wave >> 3;
    int ntile = wave & 7;
    int i0 = itile * 16, n0 = ntile * 16;
    v8f acc = {};
#pragma unroll
    for (int k0 = 0; k0 < HID; k0 += 32) {
        v16bf a = load_frag(Hgb, i0, HID, k0, lane);
        v16bf b = load_frag(Wob, n0, HID, k0, lane);
        acc = wmma_bf16(a, b, acc);
    }
    unsigned ln = lane & 15u, half = (lane >> 4) & 1u;
    int col = n0 + ln;
    float bias = bo[col];
#pragma unroll
    for (int v = 0; v < 8; ++v) {
        int row = i0 + v + (int)half * 8;
        float t = acc[v] + bias;
        y[(size_t)row * OUTD + col] = (t > 0.0f) ? t : (__expf(t) - 1.0f);
    }
}

// ---------------- launch ----------------
extern "C" void kernel_launch(void* const* d_in, const int* in_sizes, int n_in,
                              void* d_out, int out_size, void* d_ws, size_t ws_size,
                              hipStream_t stream) {
    const float* Z     = (const float*)d_in[0];   // [8,1024,256]
    const float* Asoft = (const float*)d_in[1];   // [8,1024,1024]
    const float* Wp    = (const float*)d_in[2];   // [128,256]
    const float* attn  = (const float*)d_in[3];   // [4,128]
    const float* Wo    = (const float*)d_in[4];   // [128,128]
    const float* bo    = (const float*)d_in[5];   // [128]
    float* out = (float*)d_out;

    char* w = (char*)d_ws;
    auto alloc = [&](size_t bytes) -> void* {
        void* p = (void*)w;
        w += (bytes + 255) & ~(size_t)255;
        return p;
    };
    __bf16* Zb   = (__bf16*)alloc((size_t)BK * CC * 2);        // 4 MB
    __bf16* Wpb  = (__bf16*)alloc((size_t)HID * CC * 2);
    __bf16* Wob  = (__bf16*)alloc((size_t)OUTD * HID * 2);
    float*  Hf   = (float*) alloc((size_t)BK * HID * 4);       // 4 MB
    float*  Hh   = (float*) alloc((size_t)BB * NH * KK * 4);
    float*  e    = (float*) alloc((size_t)BB * NH * KK * 4);
    __bf16* VhT  = (__bf16*)alloc((size_t)BB * NH * HID * KK * 2);  // 8 MB
    float*  dsum = (float*) alloc((size_t)BB * NH * KK * 4);
    float*  Pagg = (float*) alloc((size_t)BB * NH * KK * HID * 4);  // 16 MB
    __bf16* Hgb  = (__bf16*)alloc((size_t)BK * HID * 2);       // 2 MB

    conv_bf16_kernel<<<(BK * CC) / 256, 256, 0, stream>>>(Z, Zb, BK * CC);
    conv_bf16_kernel<<<(HID * CC) / 256, 256, 0, stream>>>(Wp, Wpb, HID * CC);
    conv_bf16_kernel<<<(OUTD * HID) / 256, 256, 0, stream>>>(Wo, Wob, OUTD * HID);

    proj_kernel<<<512, 256, 0, stream>>>(Zb, Wpb, Hf);                       // 4096 waves
    heads_kernel<<<BK / 256, 256, 0, stream>>>(Hf, attn, Hh);
    softmax_e_kernel<<<BB * NH, 256, 0, stream>>>(Hh, e);
    vht_kernel<<<(BB * NH * HID * KK) / 256, 256, 0, stream>>>(e, Hf, VhT);
    denom_kernel<<<BK, 256, 0, stream>>>(Asoft, e, dsum);
    agg_kernel<<<256, 256, 0, stream>>>(Asoft, VhT, dsum, Pagg);             // 2048 waves
    sumheads_kernel<<<(BK * HID) / 256, 256, 0, stream>>>(Pagg, Hgb);
    out_kernel<<<512, 256, 0, stream>>>(Hgb, Wob, bo, out);
}
